// BoW_81595788690059
// MI455X (gfx1250) — matrix operand order repaired and checked by reading, not compiled, not run on hardware
//
#include <hip/hip_runtime.h>
#include <hip/hip_bf16.h>
#include <math.h>

typedef _Float16 v16h __attribute__((ext_vector_type(16)));
typedef _Float16 v8h  __attribute__((ext_vector_type(8)));
typedef float    v8f  __attribute__((ext_vector_type(8)));
typedef int      v4i  __attribute__((ext_vector_type(4)));

#define B_   4
#define T_   2048
#define V_   50257
#define C_   512
#define H_   2048
#define VPAD 50304   // 393 * 128, >= V_, multiple of block-N
#define NJ   16      // T chunks for the BoW scan
#define TC   (T_ / NJ)

#if __has_builtin(__builtin_amdgcn_global_load_async_to_lds_b128)
#define ASYNC_LDS 1
#else
#define ASYNC_LDS 0
#endif

typedef __attribute__((address_space(1))) v4i* gptr_t;  // global int4*
typedef __attribute__((address_space(3))) v4i* lptr_t;  // LDS int4*

__device__ __forceinline__ void copy16_to_lds(const _Float16* g, _Float16* l) {
#if ASYNC_LDS
  __builtin_amdgcn_global_load_async_to_lds_b128((gptr_t)(void*)g, (lptr_t)(void*)l, 0, 0);
#else
  *(uint4*)l = *(const uint4*)g;
#endif
}

template <int N>
__device__ __forceinline__ void wait_async_le() {
#if ASYNC_LDS
#if __has_builtin(__builtin_amdgcn_s_wait_asynccnt)
  __builtin_amdgcn_s_wait_asynccnt(N);
#else
  if (N == 0) asm volatile("s_wait_asynccnt 0" ::: "memory");
  else        asm volatile("s_wait_asynccnt 4" ::: "memory");
#endif
#endif
}

// ---------------- transpose + convert: fp32 (K x N) -> f16 (Npad x K), zero pad rows n >= N
__global__ void transpose_to_f16(const float* __restrict__ in, _Float16* __restrict__ out,
                                 int K, int N, int Npad) {
  __shared__ float tile[32][33];
  const int nb = blockIdx.x * 32, kb = blockIdx.y * 32;
  const int tx = threadIdx.x, ty = threadIdx.y; // 32 x 8
  #pragma unroll
  for (int i = 0; i < 32; i += 8) {
    int k = kb + ty + i, n = nb + tx;
    float v = 0.f;
    if (k < K && n < N) v = in[(size_t)k * N + n];
    tile[ty + i][tx] = v;
  }
  __syncthreads();
  #pragma unroll
  for (int i = 0; i < 32; i += 8) {
    int n = nb + ty + i, k = kb + tx;
    if (n < Npad && k < K) out[(size_t)n * K + k] = (_Float16)tile[tx][ty + i];
  }
}

// ---------------- causal BoW, 3-phase scan over T ----------------
__global__ void embed_chunk_sums(const int* __restrict__ x, const float* __restrict__ wte,
                                 const float* __restrict__ wpe, float* __restrict__ cs) {
  const int b = blockIdx.x / NJ, j = blockIdx.x % NJ;
  const int c = threadIdx.x; // C_ threads
  float s = 0.f;
  for (int t = j * TC; t < (j + 1) * TC; ++t) {
    int tok = x[b * T_ + t];
    s += wte[(size_t)tok * C_ + c] + wpe[(size_t)t * C_ + c];
  }
  cs[((size_t)b * NJ + j) * C_ + c] = s;
}

__global__ void scan_chunk_sums(float* __restrict__ cs) {
  const int b = blockIdx.x, c = threadIdx.x;
  float run = 0.f;
  for (int j = 0; j < NJ; ++j) {
    size_t i = ((size_t)b * NJ + j) * C_ + c;
    float v = cs[i];
    cs[i] = run;          // exclusive prefix
    run += v;
  }
}

__global__ void embed_bow_h(const int* __restrict__ x, const float* __restrict__ wte,
                            const float* __restrict__ wpe, const float* __restrict__ cs,
                            _Float16* __restrict__ h) {
  const int b = blockIdx.x / NJ, j = blockIdx.x % NJ;
  const int c = threadIdx.x;
  float run = cs[((size_t)b * NJ + j) * C_ + c];
  for (int t = j * TC; t < (j + 1) * TC; ++t) {
    int tok = x[b * T_ + t];
    float e = wte[(size_t)tok * C_ + c] + wpe[(size_t)t * C_ + c];
    run += e;
    float hv = e + run / (float)(t + 1);   // emb + causal prefix mean
    h[((size_t)b * T_ + t) * C_ + c] = (_Float16)hv;
  }
}

// ---------------- f16 WMMA GEMM: out = A(MxK) * Bt(Npad x K)^T ----------------
// Block tile 128x128, 8 waves (4x2), 32x64 per wave, BK=32, 3-buffer async pipeline.
// EPI 0: outF[m*Nreal+n] = acc + bias[n]         (guard n < Nreal, non-temporal)
// EPI 1: outB[m*Npad +n] = f16(tanh(acc+bias[n]))
// EPI 2: outB[m*Npad +n] = f16(acc+bias[n] + (float)res[m*Npad+n])
constexpr int BM = 128, BN = 128, BK = 32;

__device__ __forceinline__ void stage_tile(const _Float16* __restrict__ A,
                                           const _Float16* __restrict__ Bt,
                                           int m0, int n0, int k0, int K,
                                           _Float16 (*As)[BK], _Float16 (*Bs)[BK], int tid) {
  #pragma unroll
  for (int i = 0; i < 2; ++i) {              // A tile: 128x32 f16 = 512 x 16B
    int idx = tid + i * 256;
    int row = idx >> 2, ch = (idx & 3) * 8;
    copy16_to_lds(A + (size_t)(m0 + row) * K + k0 + ch, &As[row][ch]);
  }
  #pragma unroll
  for (int i = 0; i < 2; ++i) {              // B tile: 128x32 f16 = 512 x 16B
    int idx = tid + i * 256;
    int row = idx >> 2, ch = (idx & 3) * 8;
    copy16_to_lds(Bt + (size_t)(n0 + row) * K + k0 + ch, &Bs[row][ch]);
  }
}

template <int EPI>
__global__ __launch_bounds__(256)
void gemm_f16_wmma(const _Float16* __restrict__ A, const _Float16* __restrict__ Bt,
                   const float* __restrict__ bias, const _Float16* __restrict__ res,
                   float* __restrict__ outF, _Float16* __restrict__ outB,
                   int M, int Npad, int K, int Nreal) {
  __shared__ __align__(16) _Float16 As[3][BM][BK];   // 3 x 8 KB
  __shared__ __align__(16) _Float16 Bs[3][BN][BK];   // 3 x 8 KB
  const int tid  = threadIdx.x;
  const int lane = tid & 31, wid = tid >> 5;         // 8 waves (wave32)
  const int wm = (wid >> 1) * 32, wn = (wid & 1) * 64;
  const int m0 = blockIdx.y * BM, n0 = blockIdx.x * BN;
  const int lm = lane & 15, lh = lane >> 4;

  const int S = K / BK;
  stage_tile(A, Bt, m0, n0, 0,  K, As[0], Bs[0], tid);
  stage_tile(A, Bt, m0, n0, BK, K, As[1], Bs[1], tid);

  v8f acc[2][4] = {};
  for (int s = 0; s < S; ++s) {
    // tile s complete (async loads finish in order; <=4 leaves only tile s+1 pending)
    if (s + 1 < S) wait_async_le<4>(); else wait_async_le<0>();
    __syncthreads();   // tile s visible everywhere; all waves done reading buf[(s-1)%3]
    if (s + 2 < S)
      stage_tile(A, Bt, m0, n0, (s + 2) * BK, K, As[(s + 2) % 3], Bs[(s + 2) % 3], tid);

    _Float16 (*Ab)[BK] = As[s % 3];
    _Float16 (*Bb)[BK] = Bs[s % 3];
    // fragments per the ISA wave32 layouts
    v16h af[2], bfr[4];
    #pragma unroll
    for (int mi = 0; mi < 2; ++mi) {
      // A 16x32: lane m=lm, kb=lh*8; elems 0-7 -> K=kb.., elems 8-15 -> K=kb+16..
      int r = wm + mi * 16 + lm;
      v8h alo = *(const v8h*)&Ab[r][lh * 8];
      v8h ahi = *(const v8h*)&Ab[r][lh * 8 + 16];
      af[mi] = __builtin_shufflevector(alo, ahi, 0,1,2,3,4,5,6,7,8,9,10,11,12,13,14,15);
    }
    #pragma unroll
    for (int ni = 0; ni < 4; ++ni) {
      // B 32x16: lane n=lm, kb=lh*8; elems 0-7 -> K=kb.., elems 8-15 -> K=kb+16..
      int n = wn + ni * 16 + lm;
      v8h blo = *(const v8h*)&Bb[n][lh * 8];
      v8h bhi = *(const v8h*)&Bb[n][lh * 8 + 16];
      bfr[ni] = __builtin_shufflevector(blo, bhi, 0,1,2,3,4,5,6,7,8,9,10,11,12,13,14,15);
    }
    #pragma unroll
    for (int mi = 0; mi < 2; ++mi)
      #pragma unroll
      for (int ni = 0; ni < 4; ++ni)
        acc[mi][ni] = __builtin_amdgcn_wmma_f32_16x16x32_f16(
            false, af[mi], false, bfr[ni], (short)0, acc[mi][ni], false, false);
  }

  // epilogue: C/D layout -> lane covers col n=lm, rows r + 8*lh
  #pragma unroll
  for (int mi = 0; mi < 2; ++mi)
    #pragma unroll
    for (int ni = 0; ni < 4; ++ni)
      #pragma unroll
      for (int r = 0; r < 8; ++r) {
        int m = m0 + wm + mi * 16 + lh * 8 + r;
        int n = n0 + wn + ni * 16 + lm;
        float v = acc[mi][ni][r];
        if (EPI == 0) {
          if (n < Nreal)
            __builtin_nontemporal_store(v + bias[n], &outF[(size_t)m * Nreal + n]);
        } else if (EPI == 1) {
          outB[(size_t)m * Npad + n] = (_Float16)tanhf(v + bias[n]);
        } else {
          outB[(size_t)m * Npad + n] =
              (_Float16)(v + bias[n] + (float)res[(size_t)m * Npad + n]);
        }
      }
}

extern "C" void kernel_launch(void* const* d_in, const int* in_sizes, int n_in,
                              void* d_out, int out_size, void* d_ws, size_t ws_size,
                              hipStream_t stream) {
  (void)in_sizes; (void)n_in; (void)out_size; (void)ws_size;
  const int*   x      = (const int*)  d_in[0];
  const float* wte    = (const float*)d_in[1];
  const float* wpe    = (const float*)d_in[2];
  const float* w_fc   = (const float*)d_in[3];
  const float* b_fc   = (const float*)d_in[4];
  const float* w_proj = (const float*)d_in[5];
  const float* b_proj = (const float*)d_in[6];
  const float* w_head = (const float*)d_in[7];
  const float* b_head = (const float*)d_in[8];
  float* logits = (float*)d_out;

  char* p = (char*)d_ws;
  auto alloc = [&](size_t bytes) { char* r = p; p += (bytes + 255) & ~(size_t)255; return r; };
  _Float16* h_f16  = (_Float16*)alloc((size_t)B_ * T_ * C_ * 2);
  _Float16* act    = (_Float16*)alloc((size_t)B_ * T_ * H_ * 2);
  _Float16* h2_f16 = (_Float16*)alloc((size_t)B_ * T_ * C_ * 2);
  _Float16* wfcT   = (_Float16*)alloc((size_t)H_ * C_ * 2);
  _Float16* wprojT = (_Float16*)alloc((size_t)C_ * H_ * 2);
  _Float16* wheadT = (_Float16*)alloc((size_t)VPAD * C_ * 2);
  float*    cs     = (float*)   alloc((size_t)B_ * NJ * C_ * 4);

  dim3 tb(32, 8);
  // w_fc (C x H) -> (H x C);  w_proj (H x C) -> (C x H);  w_head (C x V) -> (VPAD x C), zero-padded
  transpose_to_f16<<<dim3(H_ / 32,   C_ / 32), tb, 0, stream>>>(w_fc,   wfcT,   C_, H_, H_);
  transpose_to_f16<<<dim3(C_ / 32,   H_ / 32), tb, 0, stream>>>(w_proj, wprojT, H_, C_, C_);
  transpose_to_f16<<<dim3(VPAD / 32, C_ / 32), tb, 0, stream>>>(w_head, wheadT, C_, V_, VPAD);

  embed_chunk_sums<<<B_ * NJ, C_, 0, stream>>>(x, wte, wpe, cs);
  scan_chunk_sums <<<B_,      C_, 0, stream>>>(cs);
  embed_bow_h     <<<B_ * NJ, C_, 0, stream>>>(x, wte, wpe, cs, h_f16);

  const int M = B_ * T_;
  // act = tanh(h @ w_fc + b_fc)
  gemm_f16_wmma<1><<<dim3(H_ / BN, M / BM), 256, 0, stream>>>(
      h_f16, wfcT, b_fc, nullptr, nullptr, act, M, H_, C_, H_);
  // h2 = h + act @ w_proj + b_proj
  gemm_f16_wmma<2><<<dim3(C_ / BN, M / BM), 256, 0, stream>>>(
      act, wprojT, b_proj, h_f16, nullptr, h2_f16, M, C_, H_, C_);
  // logits = h2 @ w_head + b_head   (store-bandwidth bound: 1.65 GB fp32, NT stores)
  gemm_f16_wmma<0><<<dim3(VPAD / BN, M / BM), 256, 0, stream>>>(
      h2_f16, wheadT, b_head, nullptr, logits, nullptr, M, VPAD, C_, V_);
}
